// CausalConvolution_82145544503743
// MI455X (gfx1250) — matrix operand order repaired and checked by reading, not compiled, not run on hardware
//
#include <hip/hip_runtime.h>

// ---------------------------------------------------------------------------
// CausalConvolution block for MI455X (gfx1250, wave32, WMMA).
//   x   = HS @ W_in^T + b_in            GEMM1: M=16384, N=4096, K=2048
//   y   = silu(causal_dwconv(x) + c_b)  K=4 depthwise along S
//   out = y @ W_out^T + b_out           GEMM2: M=16384, N=2048, K=4096
// bf16 A/B with f32 accumulation via v_wmma_f32_16x16x32_bf16.
// Global->LDS staging uses GLOBAL_LOAD_ASYNC_TO_LDS_B128 (ASYNCcnt) to avoid
// any VGPR staging (round-1 asm showed the staging regs spilling to scratch).
// ---------------------------------------------------------------------------

typedef __attribute__((ext_vector_type(16))) __bf16 bf16x16;
typedef __attribute__((ext_vector_type(8)))  __bf16 bf16x8;
typedef __attribute__((ext_vector_type(8)))  float  f32x8;
typedef __attribute__((ext_vector_type(4)))  int    v4i;

#define BM 128
#define BN 128
#define BK 32
#define LDS_ROW 40   // bf16 elements per LDS row: 32 data + 8 pad (80 bytes)

#if defined(__gfx1250__) && __has_builtin(__builtin_amdgcn_global_load_async_to_lds_b128)
#define USE_ASYNC_LDS 1
typedef __attribute__((address_space(1))) v4i GV4;   // global int4
typedef __attribute__((address_space(3))) v4i LV4;   // LDS int4
#else
#define USE_ASYNC_LDS 0
#endif

// ---------------------------------------------------------------- conversion
__global__ __launch_bounds__(256) void cvt_f32_to_bf16(const float* __restrict__ in,
                                                       __bf16* __restrict__ out,
                                                       size_t n) {
    size_t i = (size_t)blockIdx.x * blockDim.x + threadIdx.x;
    size_t stride = (size_t)gridDim.x * blockDim.x;
    for (; i < n; i += stride) out[i] = (__bf16)in[i];
}

// ---------------------------------------------------------------- GEMM (bf16)
// C[m,n] = sum_k A[m,k] * Bw[n,k] + bias[n]
// A: M x K row-major bf16.  Bw: N x K row-major bf16 (weights as stored).
template <bool OUT_BF16>
__global__ __launch_bounds__(256) void gemm_bf16_wmma(
    const __bf16* __restrict__ A, const __bf16* __restrict__ Bw,
    const float* __restrict__ bias, void* __restrict__ Cout,
    int M, int N, int K) {

    __shared__ __align__(16) __bf16 ldsA[2][BM][LDS_ROW];
    __shared__ __align__(16) __bf16 ldsB[2][BN][LDS_ROW];

    const int tid   = threadIdx.x;
    const int lane  = tid & 31;
    const int wave  = tid >> 5;
    const int waveM = wave & 1;   // 2 waves along M  -> 64 rows each
    const int waveN = wave >> 1;  // 4 waves along N  -> 32 cols each
    const int m0 = blockIdx.y * BM;
    const int n0 = blockIdx.x * BN;

    const int rsel  = lane & 15;
    const int halfA = lane >> 4;

    f32x8 acc[4][2] = {};

    const int nk = K / BK;

    // Stage one 128x32 bf16 tile pair (A and B) for K-step kt into LDS buffer
    // `buf`. 512 x 16B chunks per tile; 256 threads -> 2 chunks each per tile.
    auto stage = [&](int kt, int buf) {
        const int k0 = kt * BK;
#pragma unroll
        for (int i = 0; i < 2; ++i) {
            const int chunk = tid + i * 256;
            const int row = chunk >> 2;            // 0..127
            const int c4  = chunk & 3;             // 16B sub-chunk in 64B row
            const __bf16* ga = A  + (size_t)(m0 + row) * K + k0 + c4 * 8;
            const __bf16* gb = Bw + (size_t)(n0 + row) * K + k0 + c4 * 8;
#if USE_ASYNC_LDS
            __builtin_amdgcn_global_load_async_to_lds_b128(
                (GV4*)ga, (LV4*)&ldsA[buf][row][c4 * 8], 0, 0);
            __builtin_amdgcn_global_load_async_to_lds_b128(
                (GV4*)gb, (LV4*)&ldsB[buf][row][c4 * 8], 0, 0);
#else
            *(uint4*)&ldsA[buf][row][c4 * 8] = *(const uint4*)ga;
            *(uint4*)&ldsB[buf][row][c4 * 8] = *(const uint4*)gb;
#endif
        }
    };
    auto wait_stage = [&]() {
#if USE_ASYNC_LDS
        asm volatile("s_wait_asynccnt 0x0" ::: "memory");
#endif
    };

    stage(0, 0);
    wait_stage();
    __syncthreads();

    for (int kt = 0; kt < nk; ++kt) {
        const int buf = kt & 1;
        if (kt + 1 < nk) stage(kt + 1, (kt + 1) & 1);   // fire-and-forget

        // ---- fragment loads from LDS (ISA 16-bit A / B VGPR layouts) ----
        bf16x16 afrag[4], bfrag[2];
#pragma unroll
        for (int i = 0; i < 4; ++i) {
            const int row = waveM * 64 + i * 16 + rsel;
            // lane<16: K 0..7 & 16..23 ; lane>=16: K 8..15 & 24..31
            bf16x8 lo = *(const bf16x8*)&ldsA[buf][row][halfA * 8];
            bf16x8 hi = *(const bf16x8*)&ldsA[buf][row][halfA * 8 + 16];
#pragma unroll
            for (int e = 0; e < 8; ++e) { afrag[i][e] = lo[e]; afrag[i][8 + e] = hi[e]; }
        }
#pragma unroll
        for (int j = 0; j < 2; ++j) {
            const int row = waveN * 32 + j * 16 + rsel;
            // lane<16: K 0..7 & 16..23 ; lane>=16: K 8..15 & 24..31
            bf16x8 lo = *(const bf16x8*)&ldsB[buf][row][halfA * 8];
            bf16x8 hi = *(const bf16x8*)&ldsB[buf][row][halfA * 8 + 16];
#pragma unroll
            for (int e = 0; e < 8; ++e) { bfrag[j][e] = lo[e]; bfrag[j][8 + e] = hi[e]; }
        }

        // ---- 8 WMMAs per wave per K-step ----
#pragma unroll
        for (int i = 0; i < 4; ++i)
#pragma unroll
            for (int j = 0; j < 2; ++j)
                acc[i][j] = __builtin_amdgcn_wmma_f32_16x16x32_bf16(
                    false, afrag[i], false, bfrag[j], (short)0, acc[i][j],
                    false, false);

        if (kt + 1 < nk) wait_stage();   // our async writes to LDS done
        __syncthreads();                 // everyone's writes visible
    }

    // ---- epilogue: C/D layout -> lane n = lane%16, m = r + 8*(lane/16) ----
    const int hi8 = halfA * 8;
#pragma unroll
    for (int j = 0; j < 2; ++j) {
        const int n = n0 + waveN * 32 + j * 16 + rsel;
        const float bv = bias[n];
#pragma unroll
        for (int i = 0; i < 4; ++i) {
            const int mbase = m0 + waveM * 64 + i * 16 + hi8;
#pragma unroll
            for (int r = 0; r < 8; ++r) {
                const float v = acc[i][j][r] + bv;
                const size_t off = (size_t)(mbase + r) * N + n;
                if (OUT_BF16) ((__bf16*)Cout)[off] = (__bf16)v;
                else          ((float*)Cout)[off]  = v;
            }
        }
    }
}

// ------------------------------------------------- causal dwconv + SiLU
// X, Y: [B, S, C] bf16.  One thread per channel, rolling 4-tap window.
#define SCHUNK 64
__global__ __launch_bounds__(256) void conv_silu_bf16(
    const __bf16* __restrict__ X, const float* __restrict__ w,
    const float* __restrict__ cb, __bf16* __restrict__ Y, int S, int C) {

    const int c  = blockIdx.x * blockDim.x + threadIdx.x;
    const int s0 = blockIdx.y * SCHUNK;
    const int b  = blockIdx.z;

    const float w0 = w[c * 4 + 0], w1 = w[c * 4 + 1];
    const float w2 = w[c * 4 + 2], w3 = w[c * 4 + 3];
    const float bias = cb[c];

    const size_t base = (size_t)b * S * (size_t)C + c;

    float xm3 = (s0 - 3 >= 0) ? (float)X[base + (size_t)(s0 - 3) * C] : 0.f;
    float xm2 = (s0 - 2 >= 0) ? (float)X[base + (size_t)(s0 - 2) * C] : 0.f;
    float xm1 = (s0 - 1 >= 0) ? (float)X[base + (size_t)(s0 - 1) * C] : 0.f;

    for (int s = s0; s < s0 + SCHUNK; ++s) {
        const float cur = (float)X[base + (size_t)s * C];
        const float v = bias + w0 * xm3 + w1 * xm2 + w2 * xm1 + w3 * cur;
        const float sv = v / (1.f + __expf(-v));   // silu
        Y[base + (size_t)s * C] = (__bf16)sv;
        xm3 = xm2; xm2 = xm1; xm1 = cur;
    }
}

// ---------------------------------------------------------------------------
extern "C" void kernel_launch(void* const* d_in, const int* in_sizes, int n_in,
                              void* d_out, int out_size, void* d_ws, size_t ws_size,
                              hipStream_t stream) {
    const float* hs     = (const float*)d_in[0];
    const float* W_in   = (const float*)d_in[1];
    const float* b_in   = (const float*)d_in[2];
    const float* conv_w = (const float*)d_in[3];
    const float* conv_b = (const float*)d_in[4];
    const float* W_out  = (const float*)d_in[5];
    const float* b_out  = (const float*)d_in[6];
    float* out = (float*)d_out;

    const int B = 4, S = 4096, H = 2048, C = 4096;
    const int M = B * S;

    // workspace layout (all bf16 staging; ~352 MB total)
    char* ws = (char*)d_ws;
    __bf16* hs_bf   = (__bf16*)ws;  ws += (size_t)M * H * sizeof(__bf16);
    __bf16* win_bf  = (__bf16*)ws;  ws += (size_t)C * H * sizeof(__bf16);
    __bf16* wout_bf = (__bf16*)ws;  ws += (size_t)H * C * sizeof(__bf16);
    __bf16* Xb      = (__bf16*)ws;  ws += (size_t)M * C * sizeof(__bf16);
    __bf16* Yb      = (__bf16*)ws;

    // fp32 -> bf16 staging
    cvt_f32_to_bf16<<<2048, 256, 0, stream>>>(hs,    hs_bf,   (size_t)M * H);
    cvt_f32_to_bf16<<<1024, 256, 0, stream>>>(W_in,  win_bf,  (size_t)C * H);
    cvt_f32_to_bf16<<<1024, 256, 0, stream>>>(W_out, wout_bf, (size_t)H * C);

    // GEMM1: x = HS @ W_in^T + b_in   -> bf16 X
    gemm_bf16_wmma<true><<<dim3(C / BN, M / BM), 256, 0, stream>>>(
        hs_bf, win_bf, b_in, (void*)Xb, M, C, H);

    // causal depthwise conv + SiLU    -> bf16 Y
    conv_silu_bf16<<<dim3(C / 256, S / SCHUNK, B), 256, 0, stream>>>(
        Xb, conv_w, conv_b, Yb, S, C);

    // GEMM2: out = Y @ W_out^T + b_out -> fp32 out
    gemm_bf16_wmma<false><<<dim3(H / BN, M / BM), 256, 0, stream>>>(
        Yb, wout_bf, b_out, (void*)out, M, H, C);
}
